// UOTHead_48576080118202
// MI455X (gfx1250) — matrix operand order repaired and checked by reading, not compiled, not run on hardware
//
#include <hip/hip_runtime.h>
#include <hip/hip_fp16.h>
#include <math.h>

typedef _Float16 half_t;
typedef __attribute__((ext_vector_type(16))) _Float16 v16h;
typedef __attribute__((ext_vector_type(8)))  _Float16 v8h;
typedef __attribute__((ext_vector_type(2)))  _Float16 v2h;
typedef __attribute__((ext_vector_type(8)))  float    v8f;
typedef __attribute__((ext_vector_type(2)))  float    v2f;

#define NPTS  4096
#define CFEAT 640
#define BPAIR 4
#define TWOB  8

// ---------------------------------------------------------------------------
// 1) per-(bb,n) sum of squares over C -> rnorm = rsqrt(ss + 1e-8)
// feats[bb][c][n] : n contiguous -> threads along n are coalesced
// ---------------------------------------------------------------------------
__global__ __launch_bounds__(256)
void rnorm_kernel(const float* __restrict__ pf, float* __restrict__ rnorm) {
    int gid = blockIdx.x * 256 + threadIdx.x;          // 0 .. 8*4096-1
    int bb = gid >> 12;
    int n  = gid & (NPTS - 1);
    const float* base = pf + (size_t)bb * CFEAT * NPTS + n;
    float ss = 0.f;
    #pragma unroll 4
    for (int c = 0; c < CFEAT; ++c) {
        float v = base[(size_t)c * NPTS];
        ss = fmaf(v, v, ss);
    }
    rnorm[gid] = rsqrtf(ss + 1e-8f);
}

// ---------------------------------------------------------------------------
// 2) transpose + normalize + fp16: fh[bb][n][c] = feats[bb][c][n] * rnorm[bb][n]
// 32x32 LDS tile transpose, block (32,8)
// ---------------------------------------------------------------------------
__global__ __launch_bounds__(256)
void to_f16_kernel(const float* __restrict__ pf, const float* __restrict__ rnorm,
                   half_t* __restrict__ fh) {
    __shared__ float tile[32][33];
    int bb = blockIdx.z;
    int nBase = blockIdx.x * 32;
    int cBase = blockIdx.y * 32;
    int tx = threadIdx.x, ty = threadIdx.y;
    #pragma unroll
    for (int k = 0; k < 4; ++k) {
        int c = cBase + ty + k * 8;
        tile[ty + k * 8][tx] = pf[((size_t)bb * CFEAT + c) * NPTS + nBase + tx];
    }
    __syncthreads();
    #pragma unroll
    for (int k = 0; k < 4; ++k) {
        int n = nBase + ty + k * 8;
        float rn = rnorm[bb * NPTS + n];
        fh[((size_t)bb * NPTS + n) * CFEAT + cBase + tx] =
            (half_t)(tile[tx][ty + k * 8] * rn);
    }
}

// ---------------------------------------------------------------------------
// 3) WMMA GEMM + fused exp:  Kh[b][n][m] = (fp16) exp((f1[n]·f2[m] - 1)/eps)
// one wave -> 64x64 tile; 20 k-steps of 32; 16 wmma per step, 16 b128 loads
// (1 load per WMMA -> ~32 f16-FLOP per L2 byte)
// ---------------------------------------------------------------------------
__device__ inline v16h load_frag_a(const half_t* p) {
    v8h lo = *(const v8h*)p;          // K = off .. off+7
    v8h hi = *(const v8h*)(p + 8);    // K = off+8 .. off+15
    v16h r;
    #pragma unroll
    for (int i = 0; i < 8; ++i) { r[i] = lo[i]; r[i + 8] = hi[i]; }
    return r;
}
__device__ inline v16h load_frag_b(const half_t* p) {
    v8h lo = *(const v8h*)p;          // K = off .. off+7
    v8h hi = *(const v8h*)(p + 8);    // K = off+8 .. off+15
    v16h r;
    #pragma unroll
    for (int i = 0; i < 8; ++i) { r[i] = lo[i]; r[i + 8] = hi[i]; }
    return r;
}

__global__ __launch_bounds__(256, 1)
void gemm_exp_kernel(const half_t* __restrict__ fh, half_t* __restrict__ Kh,
                     const float* __restrict__ epsPtr) {
    const float eps    = __expf(epsPtr[0]) + 0.03f;
    const float invEps = 1.0f / eps;

    const int wave = threadIdx.x >> 5;
    const int lane = threadIdx.x & 31;
    const int tileId = blockIdx.x * 8 + wave;      // 0..4095
    const int mBase  = (tileId >> 6) * 64;
    const int nBase  = (tileId & 63) * 64;
    const int b = blockIdx.y;

    const half_t* f1 = fh + (size_t)b * NPTS * CFEAT;
    const half_t* f2 = fh + (size_t)(BPAIR + b) * NPTS * CFEAT;

    const int l15 = lane & 15;
    const int hi  = lane >> 4;          // 0 or 1
    const int aOff = hi * 16;           // A: 16 contiguous K per lane half
    const int bOff = hi * 16;           // B: 16 contiguous K per lane half

    const half_t* pa[4];
    const half_t* pb[4];
    #pragma unroll
    for (int i = 0; i < 4; ++i) {
        pa[i] = f1 + (size_t)(mBase + i * 16 + l15) * CFEAT + aOff;
        pb[i] = f2 + (size_t)(nBase + i * 16 + l15) * CFEAT + bOff;
    }

    v8f acc[4][4];
    #pragma unroll
    for (int mi = 0; mi < 4; ++mi)
        #pragma unroll
        for (int ni = 0; ni < 4; ++ni)
            acc[mi][ni] = (v8f){};

    #pragma unroll 1
    for (int kk = 0; kk < CFEAT; kk += 32) {
        v16h af[4], bf[4];
        #pragma unroll
        for (int i = 0; i < 4; ++i) af[i] = load_frag_a(pa[i] + kk);
        #pragma unroll
        for (int i = 0; i < 4; ++i) bf[i] = load_frag_b(pb[i] + kk);
        #pragma unroll
        for (int mi = 0; mi < 4; ++mi)
            #pragma unroll
            for (int ni = 0; ni < 4; ++ni)
                acc[mi][ni] = __builtin_amdgcn_wmma_f32_16x16x32_f16(
                    false, af[mi], false, bf[ni], (short)0, acc[mi][ni], false, false);
    }

    half_t* out = Kh + (size_t)b * NPTS * NPTS;
    const int rBase = hi * 8;           // C/D: M = r + 8*(lane>=16)
    #pragma unroll
    for (int mi = 0; mi < 4; ++mi) {
        #pragma unroll
        for (int ni = 0; ni < 4; ++ni) {
            #pragma unroll
            for (int r = 0; r < 8; ++r) {
                int row = mBase + mi * 16 + r + rBase;
                int col = nBase + ni * 16 + l15;
                out[(size_t)row * NPTS + col] =
                    (half_t)__expf((acc[mi][ni][r] - 1.f) * invEps);
            }
        }
    }
}

// ---------------------------------------------------------------------------
// 4) Sinkhorn
// ---------------------------------------------------------------------------
__global__ __launch_bounds__(256)
void init_a_kernel(float* __restrict__ avec) {
    int gid = blockIdx.x * 256 + threadIdx.x;
    avec[gid] = 1.0f / NPTS;
}

// b[m] = ( (1/N) / (K^T a)[m] + 1e-8 )^power ; thread handles 2 columns
__global__ __launch_bounds__(256)
void colmv_kernel(const half_t* __restrict__ Kh, const float* __restrict__ avec,
                  float* __restrict__ bvec,
                  const float* __restrict__ gammaPtr, const float* __restrict__ epsPtr) {
    const int b  = blockIdx.y;
    const int m0 = blockIdx.x * 512 + threadIdx.x * 2;
    const half_t* Kp = Kh + (size_t)b * NPTS * NPTS + m0;
    const float*  av = avec + b * NPTS;

    __shared__ float sa[NPTS];
    for (int i = threadIdx.x; i < NPTS; i += 256) sa[i] = av[i];
    __syncthreads();

    float s0 = 0.f, s1 = 0.f;
    for (int n = 0; n < NPTS; ++n) {
        v2h kv = *(const v2h*)(Kp + (size_t)n * NPTS);
        float an = sa[n];
        s0 = fmaf((float)kv[0], an, s0);
        s1 = fmaf((float)kv[1], an, s1);
    }
    const float eps   = __expf(epsPtr[0]) + 0.03f;
    const float gam   = __expf(gammaPtr[0]);
    const float power = gam / (gam + eps);
    const float invN  = 1.0f / NPTS;
    float* bo = bvec + b * NPTS + m0;
    bo[0] = __powf(invN / (s0 + 1e-8f), power);
    bo[1] = __powf(invN / (s1 + 1e-8f), power);
}

// a[n] = ( (1/N) / (K b)[n] + 1e-8 )^power ; block per row
__global__ __launch_bounds__(256)
void rowmv_kernel(const half_t* __restrict__ Kh, const float* __restrict__ bvec,
                  float* __restrict__ avec,
                  const float* __restrict__ gammaPtr, const float* __restrict__ epsPtr) {
    const int b = blockIdx.y, n = blockIdx.x, t = threadIdx.x;
    const half_t* row = Kh + ((size_t)b * NPTS + n) * NPTS;
    const float*  bv  = bvec + b * NPTS;
    float s = 0.f;
    #pragma unroll
    for (int j = 0; j < 8; ++j) {
        int m2 = t + j * 256;                       // half2 index
        v2h kv = *(const v2h*)(row + m2 * 2);
        v2f bb = *(const v2f*)(bv + m2 * 2);
        s = fmaf((float)kv[0], bb[0], s);
        s = fmaf((float)kv[1], bb[1], s);
    }
    __shared__ float red[256];
    red[t] = s; __syncthreads();
    for (int off = 128; off > 0; off >>= 1) {
        if (t < off) red[t] += red[t + off];
        __syncthreads();
    }
    if (t == 0) {
        const float eps   = __expf(epsPtr[0]) + 0.03f;
        const float gam   = __expf(gammaPtr[0]);
        const float power = gam / (gam + eps);
        avec[b * NPTS + n] = __powf((1.0f / NPTS) / (red[0] + 1e-8f), power);
    }
}

// ---------------------------------------------------------------------------
// 5) T = a*K*b^T (fp32 out), fused row_sum and matches = (T @ coords2)/row_sum
// ---------------------------------------------------------------------------
__global__ __launch_bounds__(256)
void t_matches_kernel(const half_t* __restrict__ Kh, const float* __restrict__ avec,
                      const float* __restrict__ bvec, const float* __restrict__ coords,
                      float* __restrict__ T, float* __restrict__ matches,
                      float* __restrict__ rowsum) {
    const int b = blockIdx.y, n = blockIdx.x, t = threadIdx.x;
    const size_t rowOff = ((size_t)b * NPTS + n) * NPTS;
    const half_t* Krow = Kh + rowOff;
    const float an = avec[b * NPTS + n];
    const float* bv = bvec + b * NPTS;
    const float4* c2 = (const float4*)coords + (size_t)(BPAIR + b) * NPTS;

    float sw = 0.f, sx = 0.f, sy = 0.f, sz = 0.f;
    #pragma unroll 2
    for (int j = 0; j < 8; ++j) {
        int m = (t + j * 256) * 2;
        v2h kv = *(const v2h*)(Krow + m);
        v2f bb = *(const v2f*)(bv + m);
        float t0 = an * (float)kv[0] * bb[0];
        float t1 = an * (float)kv[1] * bb[1];
        *(v2f*)(T + rowOff + m) = (v2f){t0, t1};
        float4 ca = c2[m];
        float4 cbn = c2[m + 1];
        sw += t0 + t1;
        sx = fmaf(t0, ca.y, fmaf(t1, cbn.y, sx));
        sy = fmaf(t0, ca.z, fmaf(t1, cbn.z, sy));
        sz = fmaf(t0, ca.w, fmaf(t1, cbn.w, sz));
    }
    __shared__ float r0[256], r1[256], r2[256], r3[256];
    r0[t] = sw; r1[t] = sx; r2[t] = sy; r3[t] = sz;
    __syncthreads();
    for (int off = 128; off > 0; off >>= 1) {
        if (t < off) {
            r0[t] += r0[t + off]; r1[t] += r1[t + off];
            r2[t] += r2[t + off]; r3[t] += r3[t + off];
        }
        __syncthreads();
    }
    if (t == 0) {
        float w = r0[0];
        float inv = 1.f / (w + 1e-8f);
        float* mo = matches + ((size_t)b * NPTS + n) * 3;
        mo[0] = r1[0] * inv; mo[1] = r2[0] * inv; mo[2] = r3[0] * inv;
        rowsum[b * NPTS + n] = w;
    }
}

// ---------------------------------------------------------------------------
// 6) weighted Kabsch: covariance + 3x3 Jacobi SVD + det-corrected rotation
// ---------------------------------------------------------------------------
__global__ __launch_bounds__(256)
void rigid_kernel(const float* __restrict__ coords, const float* __restrict__ matches,
                  const float* __restrict__ rowsum, float* __restrict__ outT) {
    const int b = blockIdx.x, t = threadIdx.x;
    const float4* c1 = (const float4*)coords + (size_t)b * NPTS;
    const float*  mt = matches + (size_t)b * NPTS * 3;
    const float*  wv = rowsum + b * NPTS;

    float acc[16];
    #pragma unroll
    for (int i = 0; i < 16; ++i) acc[i] = 0.f;

    for (int n = t; n < NPTS; n += 256) {
        float w = wv[n];
        float4 c = c1[n];
        float ax = c.y, ay = c.z, az = c.w;
        float bx = mt[n * 3 + 0], by = mt[n * 3 + 1], bz = mt[n * 3 + 2];
        acc[0] += w;
        acc[1] += w * ax; acc[2] += w * ay; acc[3] += w * az;
        acc[4] += w * bx; acc[5] += w * by; acc[6] += w * bz;
        acc[7]  += w * ax * bx; acc[8]  += w * ax * by; acc[9]  += w * ax * bz;
        acc[10] += w * ay * bx; acc[11] += w * ay * by; acc[12] += w * ay * bz;
        acc[13] += w * az * bx; acc[14] += w * az * by; acc[15] += w * az * bz;
    }
    __shared__ float red[256];
    __shared__ float sums[16];
    for (int q = 0; q < 16; ++q) {
        red[t] = acc[q]; __syncthreads();
        for (int off = 128; off > 0; off >>= 1) {
            if (t < off) red[t] += red[t + off];
            __syncthreads();
        }
        if (t == 0) sums[q] = red[0];
        __syncthreads();
    }
    if (t != 0) return;

    float W = sums[0];
    float d = W + 1e-5f;
    float Sa[3] = {sums[1], sums[2], sums[3]};
    float Sb[3] = {sums[4], sums[5], sums[6]};
    float ca[3], cb[3];
    for (int i = 0; i < 3; ++i) { ca[i] = Sa[i] / d; cb[i] = Sb[i] / d; }
    float A[3][3];
    for (int i = 0; i < 3; ++i)
        for (int j = 0; j < 3; ++j)
            A[i][j] = (sums[7 + i * 3 + j] - ca[i] * Sb[j] - Sa[i] * cb[j]
                       + W * ca[i] * cb[j]) / d;

    // Jacobi eigendecomposition of S = A^T A
    float S[3][3], V[3][3] = {{1, 0, 0}, {0, 1, 0}, {0, 0, 1}};
    for (int i = 0; i < 3; ++i)
        for (int j = 0; j < 3; ++j)
            S[i][j] = A[0][i] * A[0][j] + A[1][i] * A[1][j] + A[2][i] * A[2][j];

    const int PQ[3][2] = {{0, 1}, {0, 2}, {1, 2}};
    for (int sweep = 0; sweep < 15; ++sweep) {
        for (int r = 0; r < 3; ++r) {
            int p = PQ[r][0], q = PQ[r][1];
            float apq = S[p][q];
            if (fabsf(apq) <= 1e-30f) continue;
            float tau = (S[q][q] - S[p][p]) / (2.f * apq);
            float tt = (tau >= 0.f ? 1.f : -1.f) / (fabsf(tau) + sqrtf(1.f + tau * tau));
            float cc = 1.f / sqrtf(1.f + tt * tt), ssn = tt * cc;
            for (int k = 0; k < 3; ++k) {
                float skp = S[k][p], skq = S[k][q];
                S[k][p] = cc * skp - ssn * skq;
                S[k][q] = ssn * skp + cc * skq;
            }
            for (int k = 0; k < 3; ++k) {
                float spk = S[p][k], sqk = S[q][k];
                S[p][k] = cc * spk - ssn * sqk;
                S[q][k] = ssn * spk + cc * sqk;
            }
            for (int k = 0; k < 3; ++k) {
                float vkp = V[k][p], vkq = V[k][q];
                V[k][p] = cc * vkp - ssn * vkq;
                V[k][q] = ssn * vkp + cc * vkq;
            }
        }
    }
    // sort eigenpairs descending
    float lam[3] = {S[0][0], S[1][1], S[2][2]};
    int idx[3] = {0, 1, 2};
    for (int i = 0; i < 2; ++i)
        for (int j = i + 1; j < 3; ++j)
            if (lam[idx[j]] > lam[idx[i]]) { int tmp = idx[i]; idx[i] = idx[j]; idx[j] = tmp; }
    float Vs[3][3], sig[3];
    for (int k = 0; k < 3; ++k) {
        sig[k] = sqrtf(fmaxf(lam[idx[k]], 0.f));
        for (int i = 0; i < 3; ++i) Vs[i][k] = V[i][idx[k]];
    }
    // U columns: u_k = A v_k / sigma_k (cross-product fallback for the smallest)
    float U[3][3];
    for (int k = 0; k < 3; ++k) {
        float u0 = A[0][0] * Vs[0][k] + A[0][1] * Vs[1][k] + A[0][2] * Vs[2][k];
        float u1 = A[1][0] * Vs[0][k] + A[1][1] * Vs[1][k] + A[1][2] * Vs[2][k];
        float u2 = A[2][0] * Vs[0][k] + A[2][1] * Vs[1][k] + A[2][2] * Vs[2][k];
        float nrm = sqrtf(u0 * u0 + u1 * u1 + u2 * u2);
        if (k == 2 && nrm <= 1e-6f * sig[0] + 1e-30f) {
            U[0][2] = U[1][0] * U[2][1] - U[2][0] * U[1][1];
            U[1][2] = U[2][0] * U[0][1] - U[0][0] * U[2][1];
            U[2][2] = U[0][0] * U[1][1] - U[1][0] * U[0][1];
        } else {
            float inv = 1.f / (nrm + 1e-30f);
            U[0][k] = u0 * inv; U[1][k] = u1 * inv; U[2][k] = u2 * inv;
        }
    }
    // R = Vs * U^T, with det correction (flip Vs column 2)
    float R[3][3];
    for (int i = 0; i < 3; ++i)
        for (int j = 0; j < 3; ++j)
            R[i][j] = Vs[i][0] * U[j][0] + Vs[i][1] * U[j][1] + Vs[i][2] * U[j][2];
    float det = R[0][0] * (R[1][1] * R[2][2] - R[1][2] * R[2][1])
              - R[0][1] * (R[1][0] * R[2][2] - R[1][2] * R[2][0])
              + R[0][2] * (R[1][0] * R[2][1] - R[1][1] * R[2][0]);
    if (!(det > 0.f)) {
        for (int i = 0; i < 3; ++i) Vs[i][2] = -Vs[i][2];
        for (int i = 0; i < 3; ++i)
            for (int j = 0; j < 3; ++j)
                R[i][j] = Vs[i][0] * U[j][0] + Vs[i][1] * U[j][1] + Vs[i][2] * U[j][2];
    }
    float tr[3];
    for (int i = 0; i < 3; ++i)
        tr[i] = -(R[i][0] * ca[0] + R[i][1] * ca[1] + R[i][2] * ca[2]) + cb[i];

    float* o = outT + b * 12;   // (3,4) row-major
    for (int i = 0; i < 3; ++i) {
        o[i * 4 + 0] = R[i][0]; o[i * 4 + 1] = R[i][1];
        o[i * 4 + 2] = R[i][2]; o[i * 4 + 3] = tr[i];
    }
}

// ---------------------------------------------------------------------------
// launch
// ---------------------------------------------------------------------------
extern "C" void kernel_launch(void* const* d_in, const int* in_sizes, int n_in,
                              void* d_out, int out_size, void* d_ws, size_t ws_size,
                              hipStream_t stream) {
    const float* pf    = (const float*)d_in[0];   // (8, 640, 4096, 1)
    const float* pc    = (const float*)d_in[1];   // (8*4096, 4)
    const float* gamma = (const float*)d_in[2];
    const float* epsi  = (const float*)d_in[3];
    float* out = (float*)d_out;

    const size_t T_ELEMS     = (size_t)BPAIR * NPTS * NPTS;  // 67,108,864
    const size_t MATCH_ELEMS = (size_t)BPAIR * NPTS * 3;     // 49,152

    char* ws = (char*)d_ws;
    size_t off = 0;
    float*  rnorm = (float*)(ws + off);  off += (size_t)TWOB * NPTS * 4;            // 128 KB
    half_t* fh    = (half_t*)(ws + off); off += (size_t)TWOB * NPTS * CFEAT * 2;    // 42 MB
    half_t* Kh    = (half_t*)(ws + off); off += (size_t)BPAIR * NPTS * NPTS * 2;    // 134 MB
    float*  avec  = (float*)(ws + off);  off += (size_t)BPAIR * NPTS * 4;
    float*  bvec  = (float*)(ws + off);  off += (size_t)BPAIR * NPTS * 4;
    float*  rowsum= (float*)(ws + off);  off += (size_t)BPAIR * NPTS * 4;

    // 1) norms
    rnorm_kernel<<<(TWOB * NPTS) / 256, 256, 0, stream>>>(pf, rnorm);
    // 2) normalized fp16, K-contiguous
    to_f16_kernel<<<dim3(NPTS / 32, CFEAT / 32, TWOB), dim3(32, 8), 0, stream>>>(pf, rnorm, fh);
    // 3) WMMA GEMM + exp -> Kh fp16 (L2-resident), 64x64 tile per wave
    gemm_exp_kernel<<<dim3((NPTS / 64) * (NPTS / 64) / 8, BPAIR), 256, 0, stream>>>(fh, Kh, epsi);
    // 4) Sinkhorn
    init_a_kernel<<<(BPAIR * NPTS) / 256, 256, 0, stream>>>(avec);
    for (int it = 0; it < 5; ++it) {
        colmv_kernel<<<dim3(NPTS / 512, BPAIR), 256, 0, stream>>>(Kh, avec, bvec, gamma, epsi);
        rowmv_kernel<<<dim3(NPTS, BPAIR), 256, 0, stream>>>(Kh, bvec, avec, gamma, epsi);
    }
    // 5) T + row_sum + matches
    t_matches_kernel<<<dim3(NPTS, BPAIR), 256, 0, stream>>>(
        Kh, avec, bvec, pc, out, out + T_ELEMS, rowsum);
    // 6) rigid transform
    rigid_kernel<<<BPAIR, 256, 0, stream>>>(pc, out + T_ELEMS, rowsum,
                                            out + T_ELEMS + MATCH_ELEMS);
    (void)in_sizes; (void)n_in; (void)out_size; (void)ws_size;
}
